// SelfAttention_80324478370016
// MI455X (gfx1250) — matrix operand order repaired and checked by reading, not compiled, not run on hardware
//
#include <hip/hip_runtime.h>

typedef __attribute__((ext_vector_type(16))) _Float16 v16h;
typedef __attribute__((ext_vector_type(8)))  _Float16 v8h;
typedef __attribute__((ext_vector_type(8)))  float    v8f;

#define DIM     1024
#define HEADS   16
#define DHEAD   64
#define INNER   1024
#define BATCH   2
#define SEQ     2048
#define ROWS    (BATCH*SEQ)      /* 4096 */
#define QKVCOLS (3*INNER)        /* 3072 */

static __device__ __forceinline__ v16h cat8(v8h lo, v8h hi) {
  return __builtin_shufflevector(lo, hi, 0,1,2,3,4,5,6,7,8,9,10,11,12,13,14,15);
}
static __device__ __forceinline__ v8f wmma16(v16h a, v16h b, v8f c) {
  return __builtin_amdgcn_wmma_f32_16x16x32_f16(false, a, false, b, (short)0, c, false, false);
}
// CDNA5 async global->LDS copy (16B per lane), tracked by ASYNCcnt.
static __device__ __forceinline__ void async_copy16(unsigned int lds_off, const void* gptr) {
  asm volatile("global_load_async_to_lds_b128 %0, %1, off"
               :: "v"(lds_off), "v"(gptr) : "memory");
}

// ---------------- Kernel 1: transpose + f32->f16 weight conversion ----------
__global__ void k_transpose_cvt(const float* __restrict__ in,
                                _Float16* __restrict__ out,
                                int rows, int cols) {
  int idx = blockIdx.x * blockDim.x + threadIdx.x;
  if (idx >= rows * cols) return;
  int r = idx / cols;
  int c = idx - r * cols;            // coalesced read along cols
  out[(size_t)c * rows + r] = (_Float16)in[idx];
}

// ---------------- Kernel 2: LayerNorm -> f16 --------------------------------
__global__ void k_layernorm(const float* __restrict__ x,
                            const float* __restrict__ gamma,
                            const float* __restrict__ beta,
                            _Float16* __restrict__ h) {
  __shared__ float red_s[8], red_q[8];
  const int row = blockIdx.x;
  const float* xr = x + (size_t)row * DIM;
  float v[4], s = 0.f, sq = 0.f;
#pragma unroll
  for (int i = 0; i < 4; ++i) {
    v[i] = xr[threadIdx.x + i * 256];
    s += v[i];
    sq += v[i] * v[i];
  }
#pragma unroll
  for (int off = 16; off >= 1; off >>= 1) {
    s  += __shfl_xor(s,  off, 32);
    sq += __shfl_xor(sq, off, 32);
  }
  const int w = threadIdx.x >> 5;
  if ((threadIdx.x & 31) == 0) { red_s[w] = s; red_q[w] = sq; }
  __syncthreads();
  float ts = 0.f, tq = 0.f;
#pragma unroll
  for (int i = 0; i < 8; ++i) { ts += red_s[i]; tq += red_q[i]; }
  const float mu   = ts * (1.0f / DIM);
  const float var  = tq * (1.0f / DIM) - mu * mu;
  const float rstd = rsqrtf(var + 1e-5f);
#pragma unroll
  for (int i = 0; i < 4; ++i) {
    const int c = threadIdx.x + i * 256;
    h[(size_t)row * DIM + c] = (_Float16)((v[i] - mu) * rstd * gamma[c] + beta[c]);
  }
}

// ---------------- Kernel 3: QKV GEMM (wave -> 32x64 tile) -------------------
// hN: [4096][1024] f16, wT: [3072][1024] f16 (K-contiguous per out column).
// q,k: [B,H,N,64] f16 ; vT: [B,H,64,N] f16.  Q gets *0.125 folded in.
__global__ void k_qkv_gemm(const _Float16* __restrict__ hN,
                           const _Float16* __restrict__ wT,
                           const float* __restrict__ bias,
                           _Float16* __restrict__ q,
                           _Float16* __restrict__ kk,
                           _Float16* __restrict__ vT) {
  const int lane = threadIdx.x & 31;
  const int wave = blockIdx.x * (blockDim.x >> 5) + (threadIdx.x >> 5);
  const int NT = QKVCOLS / 64;                       // 48 col tiles
  const int mt = wave / NT;
  const int nt = wave - mt * NT;
  if (mt >= ROWS / 32) return;
  const int hi = lane >> 4, ln = lane & 15;

  v8f acc[8] = {};
  const _Float16* arow0 = hN + (size_t)(mt * 32 + ln) * DIM;
  const _Float16* arow1 = arow0 + (size_t)16 * DIM;
#pragma unroll 4
  for (int k0 = 0; k0 < DIM; k0 += 32) {
    v16h a0 = cat8(*(const v8h*)(arow0 + k0 + hi * 8),
                   *(const v8h*)(arow0 + k0 + 16 + hi * 8));
    v16h a1 = cat8(*(const v8h*)(arow1 + k0 + hi * 8),
                   *(const v8h*)(arow1 + k0 + 16 + hi * 8));
#pragma unroll
    for (int t = 0; t < 4; ++t) {
      const _Float16* brow =
          wT + (size_t)(nt * 64 + t * 16 + ln) * DIM + k0 + hi * 8;
      if ((k0 & 127) == 0) __builtin_prefetch(brow + 512, 0, 0);  // global_prefetch_b8
      v16h b = cat8(*(const v8h*)brow, *(const v8h*)(brow + 16));
      acc[t]     = wmma16(a0, b, acc[t]);
      acc[4 + t] = wmma16(a1, b, acc[4 + t]);
    }
  }
#pragma unroll
  for (int rb = 0; rb < 2; ++rb) {
#pragma unroll
    for (int t = 0; t < 4; ++t) {
      const int col   = nt * 64 + t * 16 + ln;
      const float bv  = bias[col];
      const int which = col >> 10;        // 0=q 1=k 2=v
      const int inner = col & 1023;
      const int head  = inner >> 6;
      const int d     = inner & 63;
      const float sc  = (which == 0) ? 0.125f : 1.0f;
#pragma unroll
      for (int r = 0; r < 8; ++r) {
        const int row = mt * 32 + rb * 16 + hi * 8 + r;
        const int b   = row >> 11;        // /2048
        const int n   = row & 2047;
        const _Float16 hv = (_Float16)((acc[rb * 4 + t][r] + bv) * sc);
        const size_t bh = (size_t)(b * HEADS + head);
        if (which == 0)      q [(bh * SEQ + n) * DHEAD + d] = hv;
        else if (which == 1) kk[(bh * SEQ + n) * DHEAD + d] = hv;
        else                 vT[(bh * DHEAD + d) * SEQ + n] = hv;
      }
    }
  }
}

// ---------------- Kernel 4: flash attention ---------------------------------
// Block = one (b,h), 8 waves = 8 consecutive 16-row query tiles.
// K/V 32-key tiles double-buffered in LDS via async global->LDS copies.
__global__ void k_attention(const _Float16* __restrict__ q,
                            const _Float16* __restrict__ kk,
                            const _Float16* __restrict__ vT,
                            _Float16* __restrict__ o) {
  __shared__ __align__(16) _Float16 kbuf[2][32][72];  // [key][d], padded rows
  __shared__ __align__(16) _Float16 vbuf[2][64][40];  // [d][key], padded rows
  __shared__ __align__(16) _Float16 pbuf[8][16][40];  // per-wave P tile
  const int tid    = threadIdx.x;
  const int lane   = tid & 31;
  const int wlocal = tid >> 5;
  const int bh = blockIdx.x >> 4;                 // [0,32): b*16+h
  const int rt = (blockIdx.x & 15) * 8 + wlocal;  // 16-row query tile
  const int hi = lane >> 4, ln = lane & 15;

  const _Float16* qrow  = q  + ((size_t)bh * SEQ + rt * 16 + ln) * DHEAD;
  const _Float16* kbase = kk + (size_t)bh * SEQ * DHEAD;
  const _Float16* vbase = vT + (size_t)bh * DHEAD * SEQ;
  const v16h qa0 = cat8(*(const v8h*)(qrow + hi * 16),
                        *(const v8h*)(qrow + hi * 16 + 8));
  const v16h qa1 = cat8(*(const v8h*)(qrow + 32 + hi * 16),
                        *(const v8h*)(qrow + 32 + hi * 16 + 8));

  // cooperative staging: 256 threads x 16B cover each 4KB tile
  const int skey = tid >> 3, scol = (tid & 7) * 8;  // K tile [32][64]
  const int svd  = tid >> 2, svc  = (tid & 3) * 8;  // V tile [64][32]
  unsigned int klds[2], vlds[2];
#pragma unroll
  for (int p = 0; p < 2; ++p) {
    klds[p] = (unsigned int)(size_t)&kbuf[p][skey][scol];
    vlds[p] = (unsigned int)(size_t)&vbuf[p][svd][svc];
  }

  float mrow[8], lrow[8];
  v8f oacc[4] = {};
#pragma unroll
  for (int r = 0; r < 8; ++r) { mrow[r] = -1e30f; lrow[r] = 0.f; }

  // prologue: stage tile 0
  async_copy16(klds[0], kbase + (size_t)skey * DHEAD + scol);
  async_copy16(vlds[0], vbase + (size_t)svd * SEQ + svc);

  for (int it = 0; it < SEQ / 32; ++it) {
    const int kt = it * 32;
    if (it + 1 < SEQ / 32) {  // stage next tile into other buffer
      const int ktn = kt + 32;
      async_copy16(klds[(it + 1) & 1], kbase + (size_t)(ktn + skey) * DHEAD + scol);
      async_copy16(vlds[(it + 1) & 1], vbase + (size_t)svd * SEQ + ktn + svc);
      asm volatile("s_wait_asynccnt 2" ::: "memory");  // tile it done, next in flight
    } else {
      asm volatile("s_wait_asynccnt 0" ::: "memory");
    }
    __syncthreads();

    _Float16 (*KB)[72] = kbuf[it & 1];
    _Float16 (*VB)[40] = vbuf[it & 1];

    // S = Q K^T for 32 keys (two 16x16 tiles), K-dim = 64 -> 2 WMMA each
    v8f s0 = {}, s1 = {};
    {
      const v16h b00 = cat8(*(const v8h*)&KB[ln][hi * 16],
                            *(const v8h*)&KB[ln][hi * 16 + 8]);
      const v16h b01 = cat8(*(const v8h*)&KB[ln][32 + hi * 16],
                            *(const v8h*)&KB[ln][32 + hi * 16 + 8]);
      s0 = wmma16(qa0, b00, s0);
      s0 = wmma16(qa1, b01, s0);
      const v16h b10 = cat8(*(const v8h*)&KB[16 + ln][hi * 16],
                            *(const v8h*)&KB[16 + ln][hi * 16 + 8]);
      const v16h b11 = cat8(*(const v8h*)&KB[16 + ln][32 + hi * 16],
                            *(const v8h*)&KB[16 + ln][32 + hi * 16 + 8]);
      s1 = wmma16(qa0, b10, s1);
      s1 = wmma16(qa1, b11, s1);
    }
    // online softmax; row r+hi*8 lives across one 16-lane group per VGPR r
#pragma unroll
    for (int r = 0; r < 8; ++r) {
      float a = s0[r], b = s1[r];
      float mx = fmaxf(a, b);
#pragma unroll
      for (int off = 8; off >= 1; off >>= 1) mx = fmaxf(mx, __shfl_xor(mx, off, 32));
      const float mnew = fmaxf(mrow[r], mx);
      const float corr = __expf(mrow[r] - mnew);
      const float p0 = __expf(a - mnew);
      const float p1 = __expf(b - mnew);
      float rs = p0 + p1;
#pragma unroll
      for (int off = 8; off >= 1; off >>= 1) rs += __shfl_xor(rs, off, 32);
      lrow[r] = lrow[r] * corr + rs;
      mrow[r] = mnew;
#pragma unroll
      for (int t = 0; t < 4; ++t) oacc[t][r] *= corr;
      pbuf[wlocal][hi * 8 + r][ln]      = (_Float16)p0;   // C-layout -> LDS
      pbuf[wlocal][hi * 8 + r][16 + ln] = (_Float16)p1;
    }
    asm volatile("" ::: "memory");  // LDS is in-order per wave
    // reload P in A-fragment layout (row = ln, K(key) = hi*16..hi*16+15)
    const v16h pa = cat8(*(const v8h*)&pbuf[wlocal][ln][hi * 16],
                         *(const v8h*)&pbuf[wlocal][ln][hi * 16 + 8]);
    // O += P V : B frag column = d, contiguous along keys in VB
#pragma unroll
    for (int dt = 0; dt < 4; ++dt) {
      const v16h vb = cat8(*(const v8h*)&VB[dt * 16 + ln][hi * 16],
                           *(const v8h*)&VB[dt * 16 + ln][hi * 16 + 8]);
      oacc[dt] = wmma16(pa, vb, oacc[dt]);
    }
    __syncthreads();  // protect LDS tiles before next stage overwrites
  }

  const int bb = bh >> 4, hh = bh & 15;
#pragma unroll
  for (int r = 0; r < 8; ++r) {
    const float inv = 1.0f / lrow[r];
    const int n = rt * 16 + hi * 8 + r;
    _Float16* dst = o + ((size_t)bb * SEQ + n) * INNER + hh * DHEAD + ln;
#pragma unroll
    for (int dt = 0; dt < 4; ++dt)
      dst[dt * 16] = (_Float16)(oacc[dt][r] * inv);
  }
}

// ---------------- Kernel 5: output projection (wave -> 32x64, f32 out) ------
__global__ void k_out_gemm(const _Float16* __restrict__ attn,
                           const _Float16* __restrict__ wT,
                           const float* __restrict__ bias,
                           float* __restrict__ out) {
  const int lane = threadIdx.x & 31;
  const int wave = blockIdx.x * (blockDim.x >> 5) + (threadIdx.x >> 5);
  const int NT = DIM / 64;                           // 16 col tiles
  const int mt = wave / NT;
  const int nt = wave - mt * NT;
  if (mt >= ROWS / 32) return;
  const int hi = lane >> 4, ln = lane & 15;

  v8f acc[8] = {};
  const _Float16* arow0 = attn + (size_t)(mt * 32 + ln) * INNER;
  const _Float16* arow1 = arow0 + (size_t)16 * INNER;
#pragma unroll 4
  for (int k0 = 0; k0 < INNER; k0 += 32) {
    v16h a0 = cat8(*(const v8h*)(arow0 + k0 + hi * 8),
                   *(const v8h*)(arow0 + k0 + 16 + hi * 8));
    v16h a1 = cat8(*(const v8h*)(arow1 + k0 + hi * 8),
                   *(const v8h*)(arow1 + k0 + 16 + hi * 8));
#pragma unroll
    for (int t = 0; t < 4; ++t) {
      const _Float16* brow =
          wT + (size_t)(nt * 64 + t * 16 + ln) * INNER + k0 + hi * 8;
      if ((k0 & 127) == 0) __builtin_prefetch(brow + 512, 0, 0);
      v16h b = cat8(*(const v8h*)brow, *(const v8h*)(brow + 16));
      acc[t]     = wmma16(a0, b, acc[t]);
      acc[4 + t] = wmma16(a1, b, acc[4 + t]);
    }
  }
#pragma unroll
  for (int rb = 0; rb < 2; ++rb) {
#pragma unroll
    for (int t = 0; t < 4; ++t) {
      const int col  = nt * 64 + t * 16 + ln;
      const float bv = bias[col];
#pragma unroll
      for (int r = 0; r < 8; ++r) {
        const int row = mt * 32 + rb * 16 + hi * 8 + r;
        out[(size_t)row * DIM + col] = acc[rb * 4 + t][r] + bv;
      }
    }
  }
}

// ---------------- launch ----------------------------------------------------
extern "C" void kernel_launch(void* const* d_in, const int* in_sizes, int n_in,
                              void* d_out, int out_size, void* d_ws, size_t ws_size,
                              hipStream_t stream) {
  (void)in_sizes; (void)n_in; (void)out_size; (void)ws_size;
  const float* x     = (const float*)d_in[0];
  const float* gam   = (const float*)d_in[1];
  const float* bet   = (const float*)d_in[2];
  const float* w_qkv = (const float*)d_in[3];
  const float* b_qkv = (const float*)d_in[4];
  const float* w_out = (const float*)d_in[5];
  const float* b_out = (const float*)d_in[6];
  float* out = (float*)d_out;

  char* ws = (char*)d_ws;
  const size_t MB = 1u << 20;
  _Float16* hN    = (_Float16*)(ws + 0 * MB);   // 8 MB  [4096][1024]
  _Float16* wqkvT = (_Float16*)(ws + 8 * MB);   // 6 MB  [3072][1024]
  _Float16* woutT = (_Float16*)(ws + 14 * MB);  // 2 MB  [1024][1024]
  _Float16* qb    = (_Float16*)(ws + 16 * MB);  // 8 MB  [B,H,N,64]
  _Float16* kb    = (_Float16*)(ws + 24 * MB);  // 8 MB  [B,H,N,64]
  _Float16* vTb   = (_Float16*)(ws + 32 * MB);  // 8 MB  [B,H,64,N]
  _Float16* ab    = (_Float16*)(ws + 40 * MB);  // 8 MB  [B,N,1024]

  k_transpose_cvt<<<(DIM * QKVCOLS + 255) / 256, 256, 0, stream>>>(w_qkv, wqkvT, DIM, QKVCOLS);
  k_transpose_cvt<<<(INNER * DIM + 255) / 256, 256, 0, stream>>>(w_out, woutT, INNER, DIM);
  k_layernorm<<<ROWS, 256, 0, stream>>>(x, gam, bet, hN);
  // 128 row-tiles(32) * 48 col-tiles(64), 8 waves/block
  k_qkv_gemm<<<(ROWS / 32) * (QKVCOLS / 64) / 8, 256, 0, stream>>>(hN, wqkvT, b_qkv, qb, kb, vTb);
  // 32 bh * 16 blocks (8 query tiles each)
  k_attention<<<BATCH * HEADS * (SEQ / 16 / 8), 256, 0, stream>>>(qb, kb, vTb, ab);
  // 128 row-tiles(32) * 16 col-tiles(64), 8 waves/block
  k_out_gemm<<<(ROWS / 32) * (DIM / 64) / 8, 256, 0, stream>>>(ab, woutT, b_out, out);
}
